// TransducerJoint_8847632629854
// MI455X (gfx1250) — matrix operand order repaired and checked by reading, not compiled, not run on hardware
//
#include <hip/hip_runtime.h>
#include <hip/hip_fp16.h>

typedef __attribute__((ext_vector_type(16))) _Float16 v16h;
typedef __attribute__((ext_vector_type(8)))  _Float16 v8h;
typedef __attribute__((ext_vector_type(8)))  float    v8f;

__device__ __forceinline__ v8f wmma_f16(v16h a, v16h b, v8f c) {
  return __builtin_amdgcn_wmma_f32_16x16x32_f16(false, a, false, b,
                                                (short)0, c, false, false);
}

// ---------------------------------------------------------------------------
// f32 -> f16 elementwise (activations stay row-major: A-fragments load
// directly as two 16B chunks per lane).
// ---------------------------------------------------------------------------
__global__ void cvt_f32_f16(const float* __restrict__ src,
                            _Float16* __restrict__ dst, int n) {
  int i = blockIdx.x * blockDim.x + threadIdx.x;
  if (i < n) dst[i] = (_Float16)src[i];
}

// ---------------------------------------------------------------------------
// Swizzle f32 row-major weight [K][N] into WMMA f16 B-fragment order so each
// lane's v16h is one contiguous, 32B-aligned chunk:
//   dst[((kb*NT + nb)*32 + lane)*16 + idx] =
//       src[(kb*32 + (lane>>4)*8 + (idx&7) + (idx>>3)*16)*N + nb*16 + (lane&15)]
// ---------------------------------------------------------------------------
__global__ void swizzle_b(const float* __restrict__ src,
                          _Float16* __restrict__ dst, int K, int N) {
  int o = blockIdx.x * blockDim.x + threadIdx.x;
  if (o >= K * N) return;
  int NT   = N >> 4;
  int idx  = o & 15;
  int lane = (o >> 4) & 31;
  int rest = o >> 9;
  int nb   = rest % NT;
  int kb   = rest / NT;
  int sk = kb * 32 + ((lane >> 4) << 3) + (idx & 7) + ((idx >> 3) << 4);
  int sn = nb * 16 + (lane & 15);
  dst[o] = (_Float16)src[(size_t)sk * N + sn];
}

// ---------------------------------------------------------------------------
// A-fragment from row-major f16 (16-bit A 16x32 layout): lane holds
// K = hk..hk+7 in halves 0..7 and K = hk+16..hk+23 in halves 8..15.
// ---------------------------------------------------------------------------
__device__ __forceinline__ v16h load_a16(const _Float16* __restrict__ p0) {
  v8h lo = *reinterpret_cast<const v8h*>(p0);
  v8h hi = *reinterpret_cast<const v8h*>(p0 + 16);
  v16h a;
#pragma unroll
  for (int i = 0; i < 8; ++i) { a[i] = lo[i]; a[i + 8] = hi[i]; }
  return a;
}

// ---------------------------------------------------------------------------
// Pre-joint projection GEMM: C[M][ldC] = A16[M][ldA] * Bsw + bias.
// grid = (M/16, N/128), 256 threads (8 waves); wave w owns n-tile
// blockIdx.y*8 + w. K = Ksteps*32.
// ---------------------------------------------------------------------------
__global__ __launch_bounds__(256) void proj_gemm(
    const _Float16* __restrict__ A16, int ldA,
    const _Float16* __restrict__ Bsw, int NT,
    const float* __restrict__ bias,
    float* __restrict__ C, int ldC, int Ksteps) {
  int tid  = threadIdx.x;
  int lane = tid & 31;
  int w    = tid >> 5;
  int r    = lane & 15;
  int hk   = (lane >> 4) << 3;
  int m0   = blockIdx.x << 4;
  int nb   = blockIdx.y * 8 + w;

  const _Float16* pa = A16 + (size_t)(m0 + r) * ldA + hk;
  const _Float16* pb = Bsw + ((size_t)nb * 32 + lane) * 16;
  size_t bstep = (size_t)NT * 32 * 16;  // halves per kb step

  v8f acc = {};
#pragma unroll 4
  for (int kb = 0; kb < Ksteps; ++kb) {
    v16h a = load_a16(pa + kb * 32);
    v16h b = *reinterpret_cast<const v16h*>(pb + (size_t)kb * bstep);
    acc = wmma_f16(a, b, acc);
  }

  int col  = nb * 16 + (lane & 15);
  float bv = bias[col];
  int mhi  = (lane >> 4) << 3;  // C layout: lanes 16-31 hold rows M+8
#pragma unroll
  for (int j = 0; j < 8; ++j)
    C[(size_t)(m0 + mhi + j) * ldC + col] = acc[j] + bv;
}

// ---------------------------------------------------------------------------
// Fused joint kernel. Each block: fixed (b, u0), TWO t values (t0, t0+1) that
// share the same p-rows, so every W_out B-fragment feeds 2 WMMAs (halves L2
// traffic). A-tiles are stored in LDS in *fragment order* [kb][lane][16] so
// phase-2 reads are contiguous 32B per lane (2x ds_load_b128 at the 2-cycle
// bank floor; row-major layout would be a 16-way conflict).
// grid = B * (T/2) * (U/16) = 2400 blocks, 256 threads, 64 KB dynamic LDS.
// ---------------------------------------------------------------------------
__global__ __launch_bounds__(256) void joint_gemm(
    const float* __restrict__ e,        // [B*T][1024]
    const float* __restrict__ p,        // [B*U][1024]
    const _Float16* __restrict__ Wsw,   // swizzled f16 W_out, NT = 32
    const float* __restrict__ bias,     // [512]
    float* __restrict__ out) {          // [B*T*U][512]
  extern __shared__ _Float16 lds[];     // [2][32 kb][32 lane][16] = 64 KB

  const int TILE_H = 32 * 32 * 16;      // halves per A-tile

  int bid = blockIdx.x;
  int ut  = bid % 5;
  int tp  = (bid / 5) % 120;
  int b   = bid / 600;
  int t0  = tp << 1;
  int u0  = ut << 4;

  int tid = threadIdx.x;

  // ---- phase 1: two A-tiles = f16(tanh(e_row + p_row)), fragment order ----
  {
    int r = tid & 15;                   // row within tile (u offset)
    int c = tid >> 4;                   // 64-wide K chunk, 0..15
    const float4* prow = reinterpret_cast<const float4*>(
        p + (size_t)(b * 80 + u0 + r) * 1024 + c * 64);
    const float4* erow0 = reinterpret_cast<const float4*>(
        e + (size_t)(b * 240 + t0) * 1024 + c * 64);
    const float4* erow1 = reinterpret_cast<const float4*>(
        e + (size_t)(b * 240 + t0 + 1) * 1024 + c * 64);
#pragma unroll
    for (int g = 0; g < 8; ++g) {       // 8-K groups within the 64-K chunk
      float4 p0 = prow[2 * g];
      float4 p1 = prow[2 * g + 1];
      // fragment-order address for K group (c*64 + g*8):
      //   kb = c*2 + (g>>2); ko = (g&3)*8
      //   lane = r + 16*(g&1); idx = 8*((g>>1)&1)
      int kb   = c * 2 + (g >> 2);
      int lane = r + ((g & 1) << 4);
      int idx  = (g & 2) << 2;
      _Float16* dst = lds + (kb * 32 + lane) * 16 + idx;
#pragma unroll
      for (int mt = 0; mt < 2; ++mt) {
        const float4* erow = mt ? erow1 : erow0;
        float4 e0 = erow[2 * g];
        float4 e1 = erow[2 * g + 1];
        v8h h;
        h[0] = (_Float16)tanhf(e0.x + p0.x);
        h[1] = (_Float16)tanhf(e0.y + p0.y);
        h[2] = (_Float16)tanhf(e0.z + p0.z);
        h[3] = (_Float16)tanhf(e0.w + p0.w);
        h[4] = (_Float16)tanhf(e1.x + p1.x);
        h[5] = (_Float16)tanhf(e1.y + p1.y);
        h[6] = (_Float16)tanhf(e1.z + p1.z);
        h[7] = (_Float16)tanhf(e1.w + p1.w);
        *reinterpret_cast<v8h*>(dst + mt * TILE_H) = h;
      }
    }
  }
  __syncthreads();

  // ---- phase 2: 2 (M) x 4 (N) register-blocked WMMA over K=1024 ----
  int lane = tid & 31;
  int w    = tid >> 5;

  v8f acc[2][4] = {{{}, {}, {}, {}}, {{}, {}, {}, {}}};
#pragma unroll 2
  for (int kb = 0; kb < 32; ++kb) {
    const _Float16* la = lds + (kb * 32 + lane) * 16;
    v16h a0 = *reinterpret_cast<const v16h*>(la);
    v16h a1 = *reinterpret_cast<const v16h*>(la + TILE_H);
#pragma unroll
    for (int i = 0; i < 4; ++i) {
      int nb = w * 4 + i;
      v16h bf = *reinterpret_cast<const v16h*>(
          Wsw + (((size_t)kb * 32 + nb) * 32 + lane) * 16);
      acc[0][i] = wmma_f16(a0, bf, acc[0][i]);
      acc[1][i] = wmma_f16(a1, bf, acc[1][i]);
    }
  }

  // ---- phase 3: bias + store (C layout: VGPR j -> M = j + 8*(lane>=16)) ----
  int mhi = (lane >> 4) << 3;
#pragma unroll
  for (int mt = 0; mt < 2; ++mt) {
    size_t mbase = (size_t)(b * 240 + t0 + mt) * 80 + u0;
#pragma unroll
    for (int i = 0; i < 4; ++i) {
      int col  = (w * 4 + i) * 16 + (lane & 15);
      float bv = bias[col];
#pragma unroll
      for (int j = 0; j < 8; ++j)
        out[(mbase + mhi + j) * 512 + col] = acc[mt][i][j] + bv;
    }
  }
}

// ---------------------------------------------------------------------------
extern "C" void kernel_launch(void* const* d_in, const int* in_sizes, int n_in,
                              void* d_out, int out_size, void* d_ws,
                              size_t ws_size, hipStream_t stream) {
  (void)in_sizes; (void)n_in; (void)out_size; (void)ws_size;
  const float* enc   = (const float*)d_in[0];
  const float* pred  = (const float*)d_in[1];
  const float* Wenc  = (const float*)d_in[2];
  const float* benc  = (const float*)d_in[3];
  const float* Wpred = (const float*)d_in[4];
  const float* bpred = (const float*)d_in[5];
  const float* Wout  = (const float*)d_in[6];
  const float* bout  = (const float*)d_in[7];
  float* out = (float*)d_out;

  char* ws = (char*)d_ws;
  size_t off = 0;
  auto alloc = [&](size_t bytes) -> char* {
    char* ptr = ws + off;
    off += (bytes + 255) & ~(size_t)255;
    return ptr;
  };
  _Float16* enc16   = (_Float16*)alloc((size_t)960 * 640 * 2);
  _Float16* pred16  = (_Float16*)alloc((size_t)320 * 640 * 2);
  _Float16* WencSw  = (_Float16*)alloc((size_t)640 * 1024 * 2);
  _Float16* WpredSw = (_Float16*)alloc((size_t)640 * 1024 * 2);
  _Float16* WoutSw  = (_Float16*)alloc((size_t)1024 * 512 * 2);
  float*    eWs     = (float*)alloc((size_t)960 * 1024 * 4);
  float*    pWs     = (float*)alloc((size_t)320 * 1024 * 4);

  auto cdiv = [](int a, int b) { return (a + b - 1) / b; };

  cvt_f32_f16<<<cdiv(960 * 640, 256), 256, 0, stream>>>(enc, enc16, 960 * 640);
  cvt_f32_f16<<<cdiv(320 * 640, 256), 256, 0, stream>>>(pred, pred16, 320 * 640);
  swizzle_b<<<cdiv(640 * 1024, 256), 256, 0, stream>>>(Wenc, WencSw, 640, 1024);
  swizzle_b<<<cdiv(640 * 1024, 256), 256, 0, stream>>>(Wpred, WpredSw, 640, 1024);
  swizzle_b<<<cdiv(1024 * 512, 256), 256, 0, stream>>>(Wout, WoutSw, 1024, 512);

  // e = enc @ W_enc + b_enc : M=960, N=1024 (NT=64), K=640 (20 steps)
  proj_gemm<<<dim3(60, 8), 256, 0, stream>>>(enc16, 640, WencSw, 64, benc,
                                             eWs, 1024, 20);
  // p = pred @ W_pred + b_pred : M=320
  proj_gemm<<<dim3(20, 8), 256, 0, stream>>>(pred16, 640, WpredSw, 64, bpred,
                                             pWs, 1024, 20);

  // fused tanh-merge + vocab GEMM: B * T/2 * U/16 = 2400 blocks, 64 KB LDS
  joint_gemm<<<dim3(2400), 256, 65536, stream>>>(eWs, pWs, WoutSw, bout, out);
}
